// HGAT_5909875000128
// MI455X (gfx1250) — matrix operand order repaired and checked by reading, not compiled, not run on hardware
//
#include <hip/hip_runtime.h>
#include <hip/hip_bf16.h>

typedef __attribute__((ext_vector_type(4)))  _Float16 v4h;
typedef __attribute__((ext_vector_type(8)))  _Float16 v8h;
typedef __attribute__((ext_vector_type(16))) _Float16 v16h;
typedef __attribute__((ext_vector_type(4)))  float    v4f;
typedef __attribute__((ext_vector_type(8)))  float    v8f;

#define N0_ 8000
#define N1_ 6000
#define N2_ 6000
#define NN_ 20000
#define EE_ 320000
#define NH_ 128
#define NC_ 16

static inline int cdiv_i(long a, long b) { return (int)((a + b - 1) / b); }

// ---------------------------------------------------------------------------
// WMMA GEMM: C[M,Nc] = A[M,K] * B[K,Nc] (+bias), fp32 in/out, f16 compute.
// Block = 128 threads = 4 wave32s. Block tile 64x64, K step 64 (2 WMMA rounds
// per barrier). A staged row-major, B staged TRANSPOSED so every lane's WMMA
// fragment is two aligned ds_load_b128s. Row pitch 72 halves (144 B = 9*16 B)
// keeps 16-byte alignment and conflict-free bank striding (36 mod 64).
// Requires K % 4 == 0 (true for all GEMMs in this network).
// ---------------------------------------------------------------------------
__global__ __launch_bounds__(128)
void gemm_wmma_f16(const float* __restrict__ A, const float* __restrict__ B,
                   const float* __restrict__ bias, float* __restrict__ C,
                   int M, int K, int Nc, int ldc)
{
    __shared__ __attribute__((aligned(16))) _Float16 As [64][72]; // 64 rows x 64 K
    __shared__ __attribute__((aligned(16))) _Float16 BsT[64][72]; // 64 cols x 64 K

    const int tid  = threadIdx.x;
    const int wave = tid >> 5;
    const int lane = tid & 31;
    const int r    = lane & 15;   // tile-local N index (and A-frag row)
    const int sel  = lane >> 4;   // half-wave selector

    const int m0 = blockIdx.y * 64;
    const int n0 = blockIdx.x * 64;

    v8f acc[4] = {};

    for (int k0 = 0; k0 < K; k0 += 64) {
        // Prefetch next A K-tile (global_prefetch_b8 on gfx1250)
        if (k0 + 64 < K) {
            int pr = m0 + (tid >> 1);
            int pk = k0 + 64 + (tid & 1) * 32;
            if (pr < M && pk < K)
                __builtin_prefetch(A + (long)pr * K + pk, 0, 1);
        }

        // Stage A 64x64: float4 global loads -> f16, ds_store_b64 per group.
        #pragma unroll
        for (int it = 0; it < 8; ++it) {
            int g  = tid + it * 128;        // 1024 groups of 4
            int rr = g >> 4;
            int kk = (g & 15) << 2;
            int gm = m0 + rr, gk = k0 + kk;
            v4h h4;
            if (gm < M && gk + 3 < K) {     // K%4==0 -> group fully in range
                v4f f4 = *(const v4f*)(A + (long)gm * K + gk);
                h4[0] = (_Float16)f4.x; h4[1] = (_Float16)f4.y;
                h4[2] = (_Float16)f4.z; h4[3] = (_Float16)f4.w;
            } else {
                h4[0] = h4[1] = h4[2] = h4[3] = (_Float16)0.f;
            }
            *(v4h*)&As[rr][kk] = h4;
        }
        // Stage B 64x64 transposed: per-k coalesced b32 loads (consecutive
        // lanes -> consecutive columns), one ds_store_b64 per 4-k group.
        #pragma unroll
        for (int it = 0; it < 8; ++it) {
            int g  = tid + it * 128;        // 64 cols x 16 k-groups
            int cc = g & 63;
            int kk = (g >> 6) << 2;
            int gn = n0 + cc;
            v4h h4;
            #pragma unroll
            for (int j = 0; j < 4; ++j) {
                int gk = k0 + kk + j;
                float v = (gk < K && gn < Nc) ? B[(long)gk * Nc + gn] : 0.f;
                h4[j] = (_Float16)v;
            }
            *(v4h*)&BsT[cc][kk] = h4;
        }
        __syncthreads();

        const int arow = wave * 16 + r;
        #pragma unroll
        for (int kh = 0; kh < 2; ++kh) {
            // A frag: lanes 0-15: e0-7 -> K0-7, e8-15 -> K16-23; lanes 16-31: +8.
            v8h a0 = *(const v8h*)&As[arow][kh * 32 + sel * 8];
            v8h a1 = *(const v8h*)&As[arow][kh * 32 + 16 + sel * 8];
            v16h af = __builtin_shufflevector(a0, a1,
                0,1,2,3,4,5,6,7,8,9,10,11,12,13,14,15);
            #pragma unroll
            for (int t = 0; t < 4; ++t) {
                // B frag: lanes 0-15: e0-7 -> K0-7, e8-15 -> K16-23; lanes 16-31: +8.
                const int bcol = t * 16 + r;
                v8h b0 = *(const v8h*)&BsT[bcol][kh * 32 + sel * 8];
                v8h b1 = *(const v8h*)&BsT[bcol][kh * 32 + 16 + sel * 8];
                v16h bfr = __builtin_shufflevector(b0, b1,
                    0,1,2,3,4,5,6,7,8,9,10,11,12,13,14,15);
                acc[t] = __builtin_amdgcn_wmma_f32_16x16x32_f16(
                    false, af, false, bfr, (short)0, acc[t], false, false);
            }
        }
        __syncthreads();
    }

    // Epilogue. C layout: element v -> M = v + 8*sel, N = lane&15.
    #pragma unroll
    for (int t = 0; t < 4; ++t) {
        int gc = n0 + t * 16 + r;
        if (gc >= Nc) continue;
        float bv = bias ? bias[gc] : 0.0f;
        #pragma unroll
        for (int v = 0; v < 8; ++v) {
            int gr = m0 + wave * 16 + v + sel * 8;
            if (gr < M) C[(long)gr * ldc + gc] = acc[t][v] + bv;
        }
    }
}

// ---------------------------------------------------------------------------
// Elementwise / edge kernels
// ---------------------------------------------------------------------------
__global__ void fill_f32(float* p, float v, long n) {
    long i = blockIdx.x * (long)blockDim.x + threadIdx.x;
    if (i < n) p[i] = v;
}
__global__ void fill_u32(unsigned* p, unsigned v, long n) {
    long i = blockIdx.x * (long)blockDim.x + threadIdx.x;
    if (i < n) p[i] = v;
}

// el/er: per (node, head) attention logits
__global__ void eler_kernel(const float* __restrict__ feat,
                            const float* __restrict__ al,
                            const float* __restrict__ ar,
                            float* __restrict__ el, float* __restrict__ er,
                            int n, int H)
{
    long i = blockIdx.x * (long)blockDim.x + threadIdx.x;
    if (i >= (long)n * H) return;
    int nd = (int)(i / H), h = (int)(i % H);
    const float* f = feat + (long)nd * H * NH_ + (long)h * NH_;
    const float* a = al + (long)h * NH_;
    const float* b = ar + (long)h * NH_;
    float sl = 0.f, sr = 0.f;
    for (int d = 0; d < NH_; d += 4) {
        v4f fv = *(const v4f*)(f + d);
        v4f av = *(const v4f*)(a + d);
        v4f bv = *(const v4f*)(b + d);
        sl += fv.x*av.x + fv.y*av.y + fv.z*av.z + fv.w*av.w;
        sr += fv.x*bv.x + fv.y*bv.y + fv.z*bv.z + fv.w*bv.w;
    }
    el[i] = sl; er[i] = sr;
}

__device__ __forceinline__ unsigned fenc(float f) {
    unsigned u = __float_as_uint(f);
    return (u & 0x80000000u) ? ~u : (u | 0x80000000u);
}
__device__ __forceinline__ float fdec(unsigned u) {
    unsigned v = (u & 0x80000000u) ? (u & 0x7FFFFFFFu) : ~u;
    return __uint_as_float(v);
}

// pass 1: leaky_relu edge logit; segment max via monotone-uint atomicMax
__global__ void edge_pass1(const int* __restrict__ src, const int* __restrict__ dst,
                           const float* __restrict__ el, const float* __restrict__ er,
                           float* __restrict__ eraw, unsigned* __restrict__ mx,
                           long EH, int H)
{
    long i = blockIdx.x * (long)blockDim.x + threadIdx.x;
    if (i >= EH) return;
    long e = i / H; int h = (int)(i % H);
    int s = src[e], d = dst[e];
    float x = el[(long)s * H + h] + er[(long)d * H + h];
    x = (x > 0.f) ? x : 0.2f * x;
    eraw[i] = x;
    atomicMax(&mx[(long)d * H + h], fenc(x));
}

// pass 2: exp(e - max), segment-sum denominator
__global__ void edge_pass2(const int* __restrict__ dst,
                           float* __restrict__ ev, const unsigned* __restrict__ mx,
                           float* __restrict__ den, long EH, int H)
{
    long i = blockIdx.x * (long)blockDim.x + threadIdx.x;
    if (i >= EH) return;
    long e = i / H; int h = (int)(i % H);
    int d = dst[e];
    float m = fdec(mx[(long)d * H + h]);
    float v = __expf(ev[i] - m);
    ev[i] = v;
    atomicAdd(&den[(long)d * H + h], v);
}

// pass 3: one wave32 per (edge, head); 4 coalesced atomic adds per lane
__global__ void edge_pass3(const int* __restrict__ src, const int* __restrict__ dst,
                           const float* __restrict__ ev, const float* __restrict__ den,
                           const float* __restrict__ feat, float* __restrict__ agg,
                           long EH, int H)
{
    long w = (blockIdx.x * (long)blockDim.x + threadIdx.x) >> 5;
    int lane = threadIdx.x & 31;
    if (w >= EH) return;
    long e = w / H; int h = (int)(w % H);
    int s = src[e], d = dst[e];
    float dn = den[(long)d * H + h];
    dn = (dn > 1e-9f) ? dn : 1e-9f;
    float alpha = ev[w] / dn;
    const float* fs = feat + (long)s * H * NH_ + (long)h * NH_;
    float* ad = agg + (long)d * H * NH_ + (long)h * NH_;
    #pragma unroll
    for (int dd = lane; dd < NH_; dd += 32)
        atomicAdd(&ad[dd], alpha * fs[dd]);
}

// residual add + optional ELU
__global__ void post_kernel(const float* agg, const float* resid, float* out,
                            long n, int act)
{
    long i = blockIdx.x * (long)blockDim.x + threadIdx.x;
    if (i >= n) return;
    float v = agg[i];
    if (resid) v += resid[i];
    if (act) v = (v > 0.f) ? v : (__expf(v) - 1.0f);
    out[i] = v;
}

// ---------------------------------------------------------------------------
// Host orchestration
// ---------------------------------------------------------------------------
static void launch_gemm(const float* A, const float* B, const float* bias, float* C,
                        int M, int K, int Nc, int ldc, hipStream_t s)
{
    dim3 grid(cdiv_i(Nc, 64), cdiv_i(M, 64));
    gemm_wmma_f16<<<grid, dim3(128), 0, s>>>(A, B, bias, C, M, K, Nc, ldc);
}

extern "C" void kernel_launch(void* const* d_in, const int* in_sizes, int n_in,
                              void* d_out, int out_size, void* d_ws, size_t ws_size,
                              hipStream_t stream)
{
    (void)in_sizes; (void)n_in; (void)out_size; (void)ws_size;
    const float* feat0 = (const float*)d_in[0];
    const float* feat1 = (const float*)d_in[1];
    const float* feat2 = (const float*)d_in[2];
    const float* nt0   = (const float*)d_in[3];
    const float* nt1   = (const float*)d_in[4];
    const float* nt2   = (const float*)d_in[5];
    const float* Wf0 = (const float*)d_in[6];  const float* bf0 = (const float*)d_in[7];
    const float* Wf1 = (const float*)d_in[8];  const float* bf1 = (const float*)d_in[9];
    const float* Wf2 = (const float*)d_in[10]; const float* bf2 = (const float*)d_in[11];
    const float* Wn0 = (const float*)d_in[12]; const float* bn0 = (const float*)d_in[13];
    const float* Wn1 = (const float*)d_in[14]; const float* bn1 = (const float*)d_in[15];
    const float* Wn2 = (const float*)d_in[16]; const float* bn2 = (const float*)d_in[17];
    const float* Wg0 = (const float*)d_in[18];
    const float* al0 = (const float*)d_in[19]; const float* ar0 = (const float*)d_in[20];
    const float* Wg1 = (const float*)d_in[21];
    const float* al1 = (const float*)d_in[22]; const float* ar1 = (const float*)d_in[23];
    const float* Wg2 = (const float*)d_in[24];
    const float* al2 = (const float*)d_in[25]; const float* ar2 = (const float*)d_in[26];
    const float* Wres2 = (const float*)d_in[27];
    const float* Wout = (const float*)d_in[28]; const float* bout = (const float*)d_in[29];
    const int* src = (const int*)d_in[30];
    const int* dst = (const int*)d_in[31];

    // Workspace carve-up (floats)
    float* ws   = (float*)d_ws;
    float* bufA = ws;                           // N*512
    float* bufB = bufA + (long)NN_ * 512;       // N*512
    float* bufF = bufB + (long)NN_ * 512;       // N*512 (projected features)
    float* bufR = bufF + (long)NN_ * 512;       // N*128 (linear residual)
    float* el   = bufR + (long)NN_ * 128;       // N*4
    float* er   = el   + (long)NN_ * 4;         // N*4
    float* den  = er   + (long)NN_ * 4;         // N*4
    float* ee   = den  + (long)NN_ * 4;         // E*4
    unsigned* mx = (unsigned*)(ee + (long)EE_ * 4); // N*4

    // --- Input projections: h0 = [content | node-type] -> bufA [N, 256]
    launch_gemm(feat0, Wf0, bf0, bufA,                     N0_, 512, NH_, 256, stream);
    launch_gemm(feat1, Wf1, bf1, bufA + (long)N0_*256,     N1_, 256, NH_, 256, stream);
    launch_gemm(feat2, Wf2, bf2, bufA + (long)(N0_+N1_)*256, N2_, 128, NH_, 256, stream);
    launch_gemm(nt0, Wn0, bn0, bufA + NH_,                     N0_, 8, NH_, 256, stream);
    launch_gemm(nt1, Wn1, bn1, bufA + (long)N0_*256 + NH_,     N1_, 8, NH_, 256, stream);
    launch_gemm(nt2, Wn2, bn2, bufA + (long)(N0_+N1_)*256 + NH_, N2_, 8, NH_, 256, stream);

    auto gat_layer = [&](const float* hin, int indim,
                         const float* Wg, const float* al, const float* ar, int H,
                         float* agg, const float* resid, float* outp, int act) {
        const int HD = H * NH_;
        // feature projection
        launch_gemm(hin, Wg, nullptr, bufF, NN_, indim, HD, HD, stream);
        // attention logits per (node, head)
        {
            long n = (long)NN_ * H;
            eler_kernel<<<cdiv_i(n, 256), 256, 0, stream>>>(bufF, al, ar, el, er, NN_, H);
        }
        // zero accumulators (workspace is poisoned, and atomics accumulate)
        fill_f32<<<cdiv_i((long)NN_ * HD, 256), 256, 0, stream>>>(agg, 0.f, (long)NN_ * HD);
        fill_f32<<<cdiv_i((long)NN_ * H, 256), 256, 0, stream>>>(den, 0.f, (long)NN_ * H);
        fill_u32<<<cdiv_i((long)NN_ * H, 256), 256, 0, stream>>>(mx, 0u, (long)NN_ * H);
        // edge phases
        const long EH = (long)EE_ * H;
        edge_pass1<<<cdiv_i(EH, 256), 256, 0, stream>>>(src, dst, el, er, ee, mx, EH, H);
        edge_pass2<<<cdiv_i(EH, 256), 256, 0, stream>>>(dst, ee, mx, den, EH, H);
        edge_pass3<<<cdiv_i(EH * 32, 256), 256, 0, stream>>>(src, dst, ee, den, bufF, agg, EH, H);
        // residual + activation
        post_kernel<<<cdiv_i((long)NN_ * HD, 256), 256, 0, stream>>>(
            agg, resid, outp, (long)NN_ * HD, act);
    };

    // Layer 0: [N,256] -> 4 heads x 128, no residual, ELU. h1 in bufB.
    gat_layer(bufA, 2 * NH_, Wg0, al0, ar0, 4, bufB, nullptr, bufB, 1);
    // Layer 1: [N,512] -> 4 heads x 128, identity residual, ELU. h2 in bufA.
    gat_layer(bufB, 4 * NH_, Wg1, al1, ar1, 4, bufA, bufB, bufA, 1);
    // Layer 2 residual: r = h2 @ Wres2 -> bufR [N,128]
    launch_gemm(bufA, Wres2, nullptr, bufR, NN_, 4 * NH_, NH_, NH_, stream);
    // Layer 2: [N,512] -> 1 head x 128, +bufR, no act; write h straight into output
    float* hout = (float*)d_out + (long)NN_ * NC_;
    gat_layer(bufA, 4 * NH_, Wg2, al2, ar2, 1, bufB, bufR, hout, 0);
    // Classifier: logits = h @ Wout + bout -> d_out[0 : N*16]
    launch_gemm(hout, Wout, bout, (float*)d_out, NN_, NH_, NC_, NC_, stream);
}
